// HydModelNet_36962488550058
// MI455X (gfx1250) — matrix operand order repaired and checked by reading, not compiled, not run on hardware
//
#include <hip/hip_runtime.h>
#include <hip/hip_bf16.h>

typedef __attribute__((ext_vector_type(16))) _Float16 v16h;
typedef __attribute__((ext_vector_type(4)))  _Float16 v4h;
typedef __attribute__((ext_vector_type(8)))  float    v8f;

#define STEPS     730
#define BATCH     512
#define INPUT_DIM 64
#define HID       512
#define MTOT      (STEPS * BATCH)      // 373760, divisible by 64
#define M_TILE    64
#define NBLK      (MTOT / M_TILE)      // 5840

// ---------------------------------------------------------------------------
// WMMA helper (CDNA5 v_wmma_f32_16x16x32_f16, wave32)
// ---------------------------------------------------------------------------
__device__ __forceinline__ v8f wmma_f16(v16h a, v16h b, v8f c) {
    // (neg_a, A, neg_b, B, c_mod, C, reuse_a, reuse_b)
    return __builtin_amdgcn_wmma_f32_16x16x32_f16(false, a, false, b,
                                                  (short)0, c, false, false);
}

// A fragment: 16x32 (MxK), row-major source with stride ld (LDS resident).
// Lane l: M = l&15, half = l>>4. VGPR v holds K pair:
//   k0 = (v<4 ? 0 : 16) + half*8 + (v&3)*2
// Per lane this is two contiguous 8-half runs -> merges into 2x ds_load_b128.
__device__ __forceinline__ v16h load_a_frag(const _Float16* __restrict__ p, int ld) {
    const int lane = threadIdx.x & 31;
    const int m    = lane & 15;
    const int half = lane >> 4;
    const _Float16* row = p + m * ld;
    v16h a;
#pragma unroll
    for (int v = 0; v < 8; ++v) {
        int k0 = ((v < 4) ? 0 : 16) + half * 8 + (v & 3) * 2;
        a[2 * v]     = row[k0];
        a[2 * v + 1] = row[k0 + 1];
    }
    return a;
}

__device__ __forceinline__ float fast_rcp(float x) {
    return __builtin_amdgcn_rcpf(x);           // single v_rcp_f32
}

template <int ACT>   // 0 = relu, 1 = sigmoid
__device__ __forceinline__ float act_apply(float v) {
    if (ACT) return fast_rcp(1.0f + __expf(-v));
    return fmaxf(v, 0.0f);
}

// ---------------------------------------------------------------------------
// One hidden GEMM layer: (64 x K) @ (K x 512) + bias -> act -> f16 LDS out.
// 8 waves; wave w owns n-columns [w*64, w*64+64) (4 n-tiles); loops 2 m-pairs.
// Weights are pre-shuffled into B-fragment-major layout:
//   W[((ntg*ksteps + ks)*32 + lane)*16 + j]  -> single contiguous v16h per lane
// ---------------------------------------------------------------------------
template <int ACT>
__device__ __forceinline__ void layer_gemm(
    const _Float16* __restrict__ Ain, int lda, int ksteps,
    const _Float16* __restrict__ W,              // shuffled (see above)
    const float*    __restrict__ bias,           // (512)
    _Float16*       __restrict__ Hout)           // (64 x 512) f16 LDS
{
    const int tid    = threadIdx.x;
    const int wave   = tid >> 5;
    const int lane   = tid & 31;
    const int n0     = wave * 64;
    const int half   = lane >> 4;
    const int nl     = lane & 15;
    const int lane16 = lane * 16;

    for (int mp = 0; mp < 2; ++mp) {              // m-tiles {2mp, 2mp+1}
        v8f acc[2][4];
#pragma unroll
        for (int mi = 0; mi < 2; ++mi)
#pragma unroll
            for (int nt = 0; nt < 4; ++nt)
                acc[mi][nt] = (v8f){0.f,0.f,0.f,0.f,0.f,0.f,0.f,0.f};

        for (int ks = 0; ks < ksteps; ++ks) {
            v16h a0 = load_a_frag(Ain + (mp * 32)      * lda + ks * 32, lda);
            v16h a1 = load_a_frag(Ain + (mp * 32 + 16) * lda + ks * 32, lda);
            // Prefetch next k-step weight block (L2-resident stream)
            if (ks + 1 < ksteps)
                __builtin_prefetch(W + (((size_t)((wave * 4) * ksteps + ks + 1)) << 9) + lane16, 0, 1);
#pragma unroll
            for (int nt = 0; nt < 4; ++nt) {
                const _Float16* bp =
                    W + (((size_t)((wave * 4 + nt) * ksteps + ks)) << 9) + lane16;
                v16h b = *(const v16h*)bp;
                acc[0][nt] = wmma_f16(a0, b, acc[0][nt]);
                acc[1][nt] = wmma_f16(a1, b, acc[1][nt]);
            }
        }
        // Epilogue: bias + activation + f16 store (C/D layout: m = r + 8*half)
#pragma unroll
        for (int mi = 0; mi < 2; ++mi) {
#pragma unroll
            for (int nt = 0; nt < 4; ++nt) {
                int   n  = n0 + nt * 16 + nl;
                float bv = bias[n];
#pragma unroll
                for (int r = 0; r < 8; ++r) {
                    int   m = mp * 32 + mi * 16 + r + 8 * half;
                    float v = acc[mi][nt][r] + bv;
                    Hout[m * HID + n] = (_Float16)act_apply<ACT>(v);
                }
            }
        }
    }
}

// ---------------------------------------------------------------------------
// Fused 3-layer MLP over a 64-row tile.
// ACT: 0=relu, 1=sigmoid. DOSOFTMAX: in-path (relu on L2, then row softmax).
// ---------------------------------------------------------------------------
template <int ACT, int DOSOFTMAX, int C>
__global__ __launch_bounds__(256) void mlp_kernel(
    const float*    __restrict__ X,    // (MTOT, 64) f32
    const _Float16* __restrict__ W0,   // shuffled (64->512)
    const float*    __restrict__ B0,
    const _Float16* __restrict__ W1,   // shuffled (512->512)
    const float*    __restrict__ B1,
    const _Float16* __restrict__ W2,   // shuffled (512->32 padded)
    const float*    __restrict__ B2,   // (C)
    float*          __restrict__ OUT)  // (MTOT, C)
{
    __shared__ _Float16 Xs[M_TILE * INPUT_DIM];   //   8 KB
    __shared__ _Float16 H1[M_TILE * HID];         //  64 KB
    __shared__ _Float16 H2[M_TILE * HID];         //  64 KB
    __shared__ float    Z [M_TILE * 32];          //   8 KB

    const int  tid     = threadIdx.x;
    const int  wave    = tid >> 5;
    const int  lane    = tid & 31;
    const long rowBase = (long)blockIdx.x * M_TILE;

    // Stage input tile -> f16 LDS (vectorized: float4 in, 4x f16 out)
    {
        const float4* X4  = (const float4*)(X + rowBase * INPUT_DIM);
        v4h*          Xs4 = (v4h*)Xs;
        for (int i = tid; i < M_TILE * INPUT_DIM / 4; i += 256) {
            float4 x = X4[i];
            v4h h;
            h[0] = (_Float16)x.x; h[1] = (_Float16)x.y;
            h[2] = (_Float16)x.z; h[3] = (_Float16)x.w;
            Xs4[i] = h;
        }
    }
    __syncthreads();

    // Layer 0: (64x64) @ (64x512)
    layer_gemm<ACT>(Xs, INPUT_DIM, INPUT_DIM / 32, W0, B0, H1);
    __syncthreads();
    // Layer 1: (64x512) @ (512x512)
    layer_gemm<ACT>(H1, HID, HID / 32, W1, B1, H2);
    __syncthreads();

    // Layer 2: (64x512) @ (512x32pad): 8 tiles, one per wave
    {
        const int mt = wave & 3;
        const int nt = wave >> 2;
        v8f acc = (v8f){0.f,0.f,0.f,0.f,0.f,0.f,0.f,0.f};
        for (int ks = 0; ks < HID / 32; ++ks) {
            v16h a = load_a_frag(H2 + (mt * 16) * HID + ks * 32, HID);
            const _Float16* bp =
                W2 + (((size_t)(nt * (HID / 32) + ks)) << 9) + lane * 16;
            v16h b = *(const v16h*)bp;
            acc = wmma_f16(a, b, acc);
        }
        const int   half = lane >> 4;
        const int   n    = nt * 16 + (lane & 15);
        const float bv   = (n < C) ? B2[n] : 0.0f;
#pragma unroll
        for (int r = 0; r < 8; ++r) {
            int m = mt * 16 + r + 8 * half;
            Z[m * 32 + n] = acc[r] + bv;
        }
    }
    __syncthreads();

    // Per-row finish: relu+softmax (in-path) or sigmoid (out-path)
    if (tid < M_TILE) {
        const float* z = &Z[tid * 32];
        float* o = OUT + (rowBase + tid) * C;
        if (DOSOFTMAX) {
            float zr[C];
            float mx = 0.0f;                       // relu outputs are >= 0
#pragma unroll
            for (int j = 0; j < C; ++j) { zr[j] = fmaxf(z[j], 0.0f); mx = fmaxf(mx, zr[j]); }
            float s = 0.0f;
#pragma unroll
            for (int j = 0; j < C; ++j) { zr[j] = __expf(zr[j] - mx); s += zr[j]; }
            float inv = fast_rcp(s);
#pragma unroll
            for (int j = 0; j < C; ++j) o[j] = zr[j] * inv;
        } else {
#pragma unroll
            for (int j = 0; j < C; ++j) o[j] = fast_rcp(1.0f + __expf(-z[j]));
        }
    }
}

// ---------------------------------------------------------------------------
// Sequential scan: one thread per batch column, 16 stores in registers.
// ---------------------------------------------------------------------------
__global__ __launch_bounds__(256) void scan_kernel(
    const float* __restrict__ X,    // (730, 512, 64)
    const float* __restrict__ A,    // (MTOT, 17) softmax outputs
    const float* __restrict__ Bv,   // (MTOT, 16) sigmoid outputs
    float*       __restrict__ OUT)  // (730, 512)
{
    int b = blockIdx.x * 256 + threadIdx.x;
    if (b >= BATCH) return;
    float s[16];
#pragma unroll
    for (int j = 0; j < 16; ++j) s[j] = 0.0f;
    s[0] = 1000.0f;
    s[1] = 100.0f;
    for (int t = 0; t < STEPS; ++t) {
        long r = (long)t * BATCH + b;
        float rain = X[r * INPUT_DIM + 2];
        const float* a  = A  + r * 17;
        const float* bb = Bv + r * 16;
        float flow = 0.0f;
#pragma unroll
        for (int j = 0; j < 16; ++j) {
            s[j] += a[j + 1] * rain;
            float f = bb[j] * s[j];
            s[j] -= f;
            flow += f;
        }
        OUT[r] = flow;
    }
}

// ---------------------------------------------------------------------------
// Weight conversion + shuffle into WMMA B-fragment-major f16 layout.
// src: (K x N) f32 row-major.  dst: (Npad/16 n-tiles) x (K/32 k-steps) blocks,
// each block 32 lanes x 16 halfs:
//   dst[((nt*KS + ks)*32 + lane)*16 + j] = src[k , nt*16+n]
// with n = lane&15, half = lane>>4, k = ks*32 + (j>>3)*16 + half*8 + (j&7).
// Zero-pads columns >= N.
// ---------------------------------------------------------------------------
__global__ void cvt_b_shuffle(const float* __restrict__ src, _Float16* __restrict__ dst,
                              int K, int N, int Npad) {
    int i = blockIdx.x * 256 + threadIdx.x;
    int total = K * Npad;
    if (i >= total) return;
    int KS   = K >> 5;
    int j    = i & 15;
    int lane = (i >> 4) & 31;
    int blk  = i >> 9;
    int ks   = blk % KS;
    int nt   = blk / KS;
    int n    = lane & 15;
    int half = lane >> 4;
    int k    = ks * 32 + ((j >> 3) << 4) + half * 8 + (j & 7);
    int col  = nt * 16 + n;
    dst[i] = (col < N) ? (_Float16)src[k * N + col] : (_Float16)0.0f;
}

// ---------------------------------------------------------------------------
extern "C" void kernel_launch(void* const* d_in, const int* in_sizes, int n_in,
                              void* d_out, int out_size, void* d_ws, size_t ws_size,
                              hipStream_t stream) {
    const float* X   = (const float*)d_in[0];
    const float* iw0 = (const float*)d_in[1];
    const float* ib0 = (const float*)d_in[2];
    const float* iw1 = (const float*)d_in[3];
    const float* ib1 = (const float*)d_in[4];
    const float* iw2 = (const float*)d_in[5];
    const float* ib2 = (const float*)d_in[6];
    const float* ow0 = (const float*)d_in[7];
    const float* ob0 = (const float*)d_in[8];
    const float* ow1 = (const float*)d_in[9];
    const float* ob1 = (const float*)d_in[10];
    const float* ow2 = (const float*)d_in[11];
    const float* ob2 = (const float*)d_in[12];
    float* out = (float*)d_out;

    char*  ws  = (char*)d_ws;
    size_t off = 0;
    auto carve = [&](size_t bytes) -> char* {
        char* p = ws + off;
        off = (off + bytes + 255) & ~(size_t)255;
        return p;
    };
    _Float16* W0i = (_Float16*)carve((size_t)INPUT_DIM * HID * 2);
    _Float16* W1i = (_Float16*)carve((size_t)HID * HID * 2);
    _Float16* W2i = (_Float16*)carve((size_t)HID * 32 * 2);
    _Float16* W0o = (_Float16*)carve((size_t)INPUT_DIM * HID * 2);
    _Float16* W1o = (_Float16*)carve((size_t)HID * HID * 2);
    _Float16* W2o = (_Float16*)carve((size_t)HID * 32 * 2);
    float*    Aws = (float*)carve((size_t)MTOT * 17 * 4);
    float*    Bws = (float*)carve((size_t)MTOT * 16 * 4);

    // Phase 1: convert+shuffle weights into WMMA fragment layout (f16)
    cvt_b_shuffle<<<(INPUT_DIM * HID + 255) / 256, 256, 0, stream>>>(iw0, W0i, INPUT_DIM, HID, HID);
    cvt_b_shuffle<<<(HID * HID + 255) / 256,       256, 0, stream>>>(iw1, W1i, HID, HID, HID);
    cvt_b_shuffle<<<(HID * 32 + 255) / 256,        256, 0, stream>>>(iw2, W2i, HID, 17, 32);
    cvt_b_shuffle<<<(INPUT_DIM * HID + 255) / 256, 256, 0, stream>>>(ow0, W0o, INPUT_DIM, HID, HID);
    cvt_b_shuffle<<<(HID * HID + 255) / 256,       256, 0, stream>>>(ow1, W1o, HID, HID, HID);
    cvt_b_shuffle<<<(HID * 32 + 255) / 256,        256, 0, stream>>>(ow2, W2o, HID, 16, 32);

    // Phase 2: batched fused MLPs (WMMA), parallel across all T*B rows
    mlp_kernel<0, 1, 17><<<NBLK, 256, 0, stream>>>(X, W0i, ib0, W1i, ib1, W2i, ib2, Aws);
    mlp_kernel<1, 0, 16><<<NBLK, 256, 0, stream>>>(X, W0o, ob0, W1o, ob1, W2o, ob2, Bws);

    // Phase 3: sequential store-update scan (cheap, bandwidth-trivial)
    scan_kernel<<<(BATCH + 255) / 256, 256, 0, stream>>>(X, Aws, Bws, out);
}
